// MoEFFN_38165079392265
// MI455X (gfx1250) — matrix operand order repaired and checked by reading, not compiled, not run on hardware
//
#include <hip/hip_runtime.h>

// ---------------- problem constants ----------------
#define D_MODEL 1024
#define D_HID   4096
#define NE      8
#define NTOK    4096            // B*T = 4*1024
#define NSLOT   (NTOK * 2)      // top-2 assignments

typedef __attribute__((ext_vector_type(16))) __bf16 bf16x16;
typedef __attribute__((ext_vector_type(8)))  float  f32x8;

union Frag16 { bf16x16 v; uint4 q[2]; };

#if defined(__has_builtin)
#if __has_builtin(__builtin_amdgcn_global_load_async_to_lds_b128) && \
    __has_builtin(__builtin_amdgcn_s_wait_asynccnt)
#define USE_ASYNC_LDS 1
#endif
#endif

#ifdef USE_ASYNC_LDS
// ASYNCcnt-tracked direct global->LDS 16B copy (gfx1250).
__device__ __forceinline__ void async_copy_b128(const void* gsrc, void* ldst) {
    typedef int int4v __attribute__((vector_size(16)));
    __builtin_amdgcn_global_load_async_to_lds_b128(
        (__attribute__((address_space(1))) int4v*)gsrc,
        (__attribute__((address_space(3))) int4v*)ldst,
        0, 0);
}
#endif

__device__ __forceinline__ float gelu_exact(float x) {
    return 0.5f * x * (1.0f + erff(x * 0.70710678118654752f));
}

// pack two floats into one bf16x2 dword (lowers to v_cvt_pk_bf16_f32)
__device__ __forceinline__ unsigned pack_bf16(float lo, float hi) {
    union { __bf16 h[2]; unsigned u; } t;
    t.h[0] = (__bf16)lo;
    t.h[1] = (__bf16)hi;
    return t.u;
}

// WMMA compute step: 8 waves arranged 4(M) x 2(N); each wave owns a 32x64
// sub-tile = 2x4 fragments of 16x16, K-step = 32 (bf16).
// As: [128][40] bf16 (M-major), Bs: [128][40] bf16 (N-major: per-column K runs)
__device__ __forceinline__ void wmma_step(const __bf16 (*As)[40],
                                          const __bf16 (*Bs)[40],
                                          f32x8 acc[2][4],
                                          int wm, int wn, int l15, int half) {
    Frag16 fa[2], fb[4];
#pragma unroll
    for (int fm = 0; fm < 2; ++fm) {
        int m = wm * 32 + fm * 16 + l15;
        fa[fm].q[0] = *reinterpret_cast<const uint4*>(&As[m][8 * half]);        // K = 8h..8h+7
        fa[fm].q[1] = *reinterpret_cast<const uint4*>(&As[m][16 + 8 * half]);   // K = 16+8h..+7
    }
#pragma unroll
    for (int fn = 0; fn < 4; ++fn) {
        int n = wn * 64 + fn * 16 + l15;
        fb[fn].q[0] = *reinterpret_cast<const uint4*>(&Bs[n][8 * half]);        // K = 8h..8h+7
        fb[fn].q[1] = *reinterpret_cast<const uint4*>(&Bs[n][16 + 8 * half]);   // K = 16+8h..+7
    }
#pragma unroll
    for (int fm = 0; fm < 2; ++fm)
#pragma unroll
        for (int fn = 0; fn < 4; ++fn)
            acc[fm][fn] = __builtin_amdgcn_wmma_f32_16x16x32_bf16(
                false, fa[fm].v, false, fb[fn].v, (short)0, acc[fm][fn], false, false);
}

// convert 4x float4 -> 16 bf16 row segment in LDS (A tile, M-major; compiler
// merges these into cvt_pk + ds_store_b128)
__device__ __forceinline__ void store_a_bf16(__bf16 (*As)[40], int row, int cb,
                                             const float4 v[4]) {
#pragma unroll
    for (int i = 0; i < 4; ++i) {
        As[row][cb + i * 4 + 0] = (__bf16)v[i].x;
        As[row][cb + i * 4 + 1] = (__bf16)v[i].y;
        As[row][cb + i * 4 + 2] = (__bf16)v[i].z;
        As[row][cb + i * 4 + 3] = (__bf16)v[i].w;
    }
}

// B transpose store: two adjacent K-rows x 8 N-cols -> 8 packed b32 LDS stores.
// r0 = rows {k2}, r1 = rows {k2+1}; each pack is one v_cvt_pk_bf16_f32.
__device__ __forceinline__ void store_b_pairs(__bf16 (*Bs)[40], int nb8, int k2,
                                              const float4& r0a, const float4& r0b,
                                              const float4& r1a, const float4& r1b) {
    *reinterpret_cast<unsigned*>(&Bs[nb8 + 0][k2]) = pack_bf16(r0a.x, r1a.x);
    *reinterpret_cast<unsigned*>(&Bs[nb8 + 1][k2]) = pack_bf16(r0a.y, r1a.y);
    *reinterpret_cast<unsigned*>(&Bs[nb8 + 2][k2]) = pack_bf16(r0a.z, r1a.z);
    *reinterpret_cast<unsigned*>(&Bs[nb8 + 3][k2]) = pack_bf16(r0a.w, r1a.w);
    *reinterpret_cast<unsigned*>(&Bs[nb8 + 4][k2]) = pack_bf16(r0b.x, r1b.x);
    *reinterpret_cast<unsigned*>(&Bs[nb8 + 5][k2]) = pack_bf16(r0b.y, r1b.y);
    *reinterpret_cast<unsigned*>(&Bs[nb8 + 6][k2]) = pack_bf16(r0b.z, r1b.z);
    *reinterpret_cast<unsigned*>(&Bs[nb8 + 7][k2]) = pack_bf16(r0b.w, r1b.w);
}

// ---------------- kernel 0: zero expert counters ----------------
__global__ void moe_zero_counts(int* counts) {
    if (threadIdx.x < NE) counts[threadIdx.x] = 0;
}

// ---------------- kernel 1: router (softmax + top-2 + expert lists) --------
__global__ void moe_router(const float* __restrict__ x,
                           const float* __restrict__ gate_w,
                           int* __restrict__ counts,
                           float* __restrict__ top_w,   // [NSLOT]
                           int* __restrict__ etok)      // [NE][NTOK], stores slot ids
{
    __shared__ __align__(16) float gw[NE * D_MODEL];    // 32 KB
    const int tid = threadIdx.x;
    for (int i = tid; i < (NE * D_MODEL) / 4; i += 256)
        reinterpret_cast<float4*>(gw)[i] = reinterpret_cast<const float4*>(gate_w)[i];
    __syncthreads();

    const int t = blockIdx.x * 256 + tid;
    float a[NE];
#pragma unroll
    for (int e = 0; e < NE; ++e) a[e] = 0.0f;

    const float4* xr = reinterpret_cast<const float4*>(x + (size_t)t * D_MODEL);
    for (int d4 = 0; d4 < D_MODEL / 4; ++d4) {
        float4 xv = xr[d4];
#pragma unroll
        for (int e = 0; e < NE; ++e) {
            float4 gv = *reinterpret_cast<const float4*>(&gw[e * D_MODEL + d4 * 4]);
            a[e] += xv.x * gv.x + xv.y * gv.y + xv.z * gv.z + xv.w * gv.w;
        }
    }
    float mx = a[0];
#pragma unroll
    for (int e = 1; e < NE; ++e) mx = fmaxf(mx, a[e]);
    float s = 0.0f, sc[NE];
#pragma unroll
    for (int e = 0; e < NE; ++e) { sc[e] = expf(a[e] - mx); s += sc[e]; }
    float inv = 1.0f / s;
#pragma unroll
    for (int e = 0; e < NE; ++e) sc[e] *= inv;
    int i0 = 0;
#pragma unroll
    for (int e = 1; e < NE; ++e) if (sc[e] > sc[i0]) i0 = e;
    int i1 = (i0 == 0) ? 1 : 0;
#pragma unroll
    for (int e = 0; e < NE; ++e) if (e != i0 && sc[e] > sc[i1]) i1 = e;
    float w0 = sc[i0], w1 = sc[i1];
    float wn = 1.0f / (w0 + w1);
    w0 *= wn; w1 *= wn;

    top_w[2 * t]     = w0;
    top_w[2 * t + 1] = w1;
    int p0 = atomicAdd(&counts[i0], 1);
    etok[i0 * NTOK + p0] = 2 * t;
    int p1 = atomicAdd(&counts[i1], 1);
    etok[i1 * NTOK + p1] = 2 * t + 1;
}

// ---------------- kernel 2: expert GEMM1 + GELU  (h = gelu(x @ w1 + b1)) ---
// grid.x = NE * 32(Mtiles) * 32(Ntiles), block = 256 (8 wave32)
// Double-buffered LDS: one barrier per K-step; next tile loads overlap WMMA.
__global__ void __launch_bounds__(256)
moe_ffn1(const float* __restrict__ x,
         const float* __restrict__ w1,
         const float* __restrict__ b1,
         const int* __restrict__ counts,
         const int* __restrict__ etok,
         __bf16* __restrict__ Hbuf)     // [NSLOT][D_HID]
{
    const int b  = blockIdx.x;
    const int e  = b >> 10;             // / (32*32)
    const int mt = (b >> 5) & 31;
    const int nt = b & 31;
    const int cnt = counts[e];
    if (mt * 128 >= cnt) return;

    __shared__ __align__(16) __bf16 As[2][128][40];    // 2 x 10 KB
    __shared__ __align__(16) __bf16 Bs[2][128][40];    // 2 x 10 KB
    __shared__ int rowslot[128];

    const int tid = threadIdx.x;
    if (tid < 128) {
        int r = mt * 128 + tid;
        rowslot[tid] = (r < cnt) ? etok[e * NTOK + r] : -1;
    }
    __syncthreads();

    const int lane = tid & 31, wave = tid >> 5;
    const int wm = wave >> 1, wn = wave & 1;
    const int l15 = lane & 15, half = lane >> 4;
    const int n0 = nt * 128;
    const float* w1e = w1 + (size_t)e * D_MODEL * D_HID;

    f32x8 acc[2][4];
#pragma unroll
    for (int i = 0; i < 2; ++i)
#pragma unroll
        for (int j = 0; j < 4; ++j)
#pragma unroll
            for (int k = 0; k < 8; ++k) acc[i][j][k] = 0.0f;

    // A-load mapping: row = tid>>1, cols (tid&1)*16 .. +15.
    // Padded rows read token 0: garbage only pollutes C rows that are never stored.
    const int arow = tid >> 1, acb = (tid & 1) * 16;
    const int aslot = rowslot[arow];
    const int atok  = (aslot >= 0) ? (aslot >> 1) : 0;
    const float* asrc = x + (size_t)atok * D_MODEL + acb;
    // B-load mapping: two adjacent K rows (k2, k2+1) x 8 N cols per thread
    const int k2  = (tid & 15) * 2;
    const int nb8 = (tid >> 4) * 8;
    const float* bsrc = w1e + (size_t)k2 * D_HID + n0 + nb8;

    constexpr int KT = D_MODEL / 32;

    // prologue: tile 0 -> buffer 0
    {
        float4 av[4];
#pragma unroll
        for (int i = 0; i < 4; ++i) av[i] = *reinterpret_cast<const float4*>(asrc + i * 4);
        float4 r0a = *reinterpret_cast<const float4*>(bsrc);
        float4 r0b = *reinterpret_cast<const float4*>(bsrc + 4);
        float4 r1a = *reinterpret_cast<const float4*>(bsrc + D_HID);
        float4 r1b = *reinterpret_cast<const float4*>(bsrc + D_HID + 4);
        store_a_bf16(As[0], arow, acb, av);
        store_b_pairs(Bs[0], nb8, k2, r0a, r0b, r1a, r1b);
    }
    __syncthreads();

    for (int kt = 0; kt < KT; ++kt) {
        const int cur = kt & 1, nxt = cur ^ 1;
        float4 av[4], r0a, r0b, r1a, r1b;
        const bool more = (kt + 1 < KT);
        if (more) {    // issue next tile's global loads; latency hides under WMMA
            const float* an = asrc + (kt + 1) * 32;
            const float* bn = bsrc + (size_t)(kt + 1) * 32 * D_HID;
#pragma unroll
            for (int i = 0; i < 4; ++i) av[i] = *reinterpret_cast<const float4*>(an + i * 4);
            r0a = *reinterpret_cast<const float4*>(bn);
            r0b = *reinterpret_cast<const float4*>(bn + 4);
            r1a = *reinterpret_cast<const float4*>(bn + D_HID);
            r1b = *reinterpret_cast<const float4*>(bn + D_HID + 4);
            if (kt + 2 < KT)   // global_prefetch_b8 two tiles ahead
                __builtin_prefetch(bsrc + (size_t)(kt + 2) * 32 * D_HID, 0, 3);
        }
        wmma_step(As[cur], Bs[cur], acc, wm, wn, l15, half);
        if (more) {
            store_a_bf16(As[nxt], arow, acb, av);
            store_b_pairs(Bs[nxt], nb8, k2, r0a, r0b, r1a, r1b);
        }
        __syncthreads();
    }

    // epilogue: +b1, exact GELU, bf16 store (C layout: m = j + 8*half, n = l15)
#pragma unroll
    for (int fm = 0; fm < 2; ++fm)
#pragma unroll
        for (int fn = 0; fn < 4; ++fn) {
            const int col = n0 + wn * 64 + fn * 16 + l15;
            const float bias = b1[e * D_HID + col];
#pragma unroll
            for (int j = 0; j < 8; ++j) {
                const int lr = wm * 32 + fm * 16 + half * 8 + j;
                const int slot = rowslot[lr];
                if (slot < 0) continue;
                float v = acc[fm][fn][j] + bias;
                Hbuf[(size_t)slot * D_HID + col] = (__bf16)gelu_exact(v);
            }
        }
}

// ---------------- kernel 3: expert GEMM2  (y = h @ w2 + b2) ----------------
// grid.x = NE * 32(Mtiles) * 8(Ntiles), block = 256. A tile is raw bf16 ->
// GLOBAL_LOAD_ASYNC_TO_LDS_B128 (ASYNCcnt) when the toolchain exposes it.
__global__ void __launch_bounds__(256)
moe_ffn2(const __bf16* __restrict__ Hbuf,
         const float* __restrict__ w2,
         const float* __restrict__ b2,
         const int* __restrict__ counts,
         const int* __restrict__ etok,
         float* __restrict__ yslot)     // [NSLOT][D_MODEL]
{
    const int b  = blockIdx.x;
    const int e  = b >> 8;              // / (32*8)
    const int mt = (b >> 3) & 31;
    const int nt = b & 7;
    const int cnt = counts[e];
    if (mt * 128 >= cnt) return;

    __shared__ __align__(16) __bf16 As[2][128][40];
    __shared__ __align__(16) __bf16 Bs[2][128][40];
    __shared__ int rowslot[128];

    const int tid = threadIdx.x;
    if (tid < 128) {
        int r = mt * 128 + tid;
        rowslot[tid] = (r < cnt) ? etok[e * NTOK + r] : -1;
    }
    __syncthreads();

    const int lane = tid & 31, wave = tid >> 5;
    const int wm = wave >> 1, wn = wave & 1;
    const int l15 = lane & 15, half = lane >> 4;
    const int n0 = nt * 128;
    const float* w2e = w2 + (size_t)e * D_HID * D_MODEL;

    f32x8 acc[2][4];
#pragma unroll
    for (int i = 0; i < 2; ++i)
#pragma unroll
        for (int j = 0; j < 4; ++j)
#pragma unroll
            for (int k = 0; k < 8; ++k) acc[i][j][k] = 0.0f;

    const int arow = tid >> 1, acb = (tid & 1) * 16;   // 16 bf16 cols per thread
    const int aslot = rowslot[arow];
    const int srcslot = (aslot >= 0) ? aslot : rowslot[0];  // every slot row is valid
    const __bf16* asrc = Hbuf + (size_t)srcslot * D_HID + acb;
    const int k2  = (tid & 15) * 2;
    const int nb8 = (tid >> 4) * 8;
    const float* bsrc = w2e + (size_t)k2 * D_MODEL + n0 + nb8;

    constexpr int KT = D_HID / 32;

    // prologue: tile 0 -> buffer 0
    {
#ifdef USE_ASYNC_LDS
        async_copy_b128(asrc,     &As[0][arow][acb]);
        async_copy_b128(asrc + 8, &As[0][arow][acb + 8]);
#else
        const uint4* s = reinterpret_cast<const uint4*>(asrc);
        *reinterpret_cast<uint4*>(&As[0][arow][acb])     = s[0];
        *reinterpret_cast<uint4*>(&As[0][arow][acb + 8]) = s[1];
#endif
        float4 r0a = *reinterpret_cast<const float4*>(bsrc);
        float4 r0b = *reinterpret_cast<const float4*>(bsrc + 4);
        float4 r1a = *reinterpret_cast<const float4*>(bsrc + D_MODEL);
        float4 r1b = *reinterpret_cast<const float4*>(bsrc + D_MODEL + 4);
        store_b_pairs(Bs[0], nb8, k2, r0a, r0b, r1a, r1b);
#ifdef USE_ASYNC_LDS
        __builtin_amdgcn_s_wait_asynccnt(0);
#endif
    }
    __syncthreads();

    for (int kt = 0; kt < KT; ++kt) {
        const int cur = kt & 1, nxt = cur ^ 1;
        float4 r0a, r0b, r1a, r1b;
        const bool more = (kt + 1 < KT);
#ifndef USE_ASYNC_LDS
        uint4 q0, q1;
#endif
        if (more) {
            const __bf16* an = asrc + (kt + 1) * 32;
#ifdef USE_ASYNC_LDS
            async_copy_b128(an,     &As[nxt][arow][acb]);
            async_copy_b128(an + 8, &As[nxt][arow][acb + 8]);
#else
            const uint4* s = reinterpret_cast<const uint4*>(an);
            q0 = s[0]; q1 = s[1];
#endif
            const float* bn = bsrc + (size_t)(kt + 1) * 32 * D_MODEL;
            r0a = *reinterpret_cast<const float4*>(bn);
            r0b = *reinterpret_cast<const float4*>(bn + 4);
            r1a = *reinterpret_cast<const float4*>(bn + D_MODEL);
            r1b = *reinterpret_cast<const float4*>(bn + D_MODEL + 4);
            if (kt + 2 < KT)
                __builtin_prefetch(bsrc + (size_t)(kt + 2) * 32 * D_MODEL, 0, 3);
        }
        wmma_step(As[cur], Bs[cur], acc, wm, wn, l15, half);
        if (more) {
#ifndef USE_ASYNC_LDS
            *reinterpret_cast<uint4*>(&As[nxt][arow][acb])     = q0;
            *reinterpret_cast<uint4*>(&As[nxt][arow][acb + 8]) = q1;
#endif
            store_b_pairs(Bs[nxt], nb8, k2, r0a, r0b, r1a, r1b);
#ifdef USE_ASYNC_LDS
            __builtin_amdgcn_s_wait_asynccnt(0);
#endif
        }
        __syncthreads();
    }

#pragma unroll
    for (int fm = 0; fm < 2; ++fm)
#pragma unroll
        for (int fn = 0; fn < 4; ++fn) {
            const int col = n0 + wn * 64 + fn * 16 + l15;
            const float bias = b2[e * D_MODEL + col];
#pragma unroll
            for (int j = 0; j < 8; ++j) {
                const int lr = wm * 32 + fm * 16 + half * 8 + j;
                const int slot = rowslot[lr];
                if (slot < 0) continue;
                yslot[(size_t)slot * D_MODEL + col] = acc[fm][fn][j] + bias;
            }
        }
}

// ---------------- kernel 4: deterministic weighted combine ----------------
__global__ void moe_combine(const float* __restrict__ yslot,
                            const float* __restrict__ top_w,
                            float* __restrict__ out)
{
    const int t = blockIdx.x;
    const int d = threadIdx.x * 4;
    const float w0 = top_w[2 * t], w1 = top_w[2 * t + 1];
    float4 y0 = *reinterpret_cast<const float4*>(yslot + (size_t)(2 * t)     * D_MODEL + d);
    float4 y1 = *reinterpret_cast<const float4*>(yslot + (size_t)(2 * t + 1) * D_MODEL + d);
    float4 o;
    o.x = w0 * y0.x + w1 * y1.x;
    o.y = w0 * y0.y + w1 * y1.y;
    o.z = w0 * y0.z + w1 * y1.z;
    o.w = w0 * y0.w + w1 * y1.w;
    *reinterpret_cast<float4*>(out + (size_t)t * D_MODEL + d) = o;
}

// ---------------- host launcher ----------------
extern "C" void kernel_launch(void* const* d_in, const int* in_sizes, int n_in,
                              void* d_out, int out_size, void* d_ws, size_t ws_size,
                              hipStream_t stream) {
    const float* x      = (const float*)d_in[0];   // [4,1024,1024]
    const float* gate_w = (const float*)d_in[1];   // [8,1024]
    const float* w1     = (const float*)d_in[2];   // [8,1024,4096]
    const float* b1     = (const float*)d_in[3];   // [8,1,4096]
    const float* w2     = (const float*)d_in[4];   // [8,4096,1024]
    const float* b2     = (const float*)d_in[5];   // [8,1,1024]
    float* out = (float*)d_out;

    // workspace carve-up (~96.2 MB total)
    char* p = (char*)d_ws;
    int*    counts = (int*)p;    p += 256;
    float*  top_w  = (float*)p;  p += (size_t)NSLOT * sizeof(float);                 // 32 KB
    int*    etok   = (int*)p;    p += (size_t)NE * NTOK * sizeof(int);               // 128 KB
    __bf16* Hbuf   = (__bf16*)p; p += (size_t)NSLOT * D_HID * sizeof(__bf16);        // 64 MB
    float*  yslot  = (float*)p;  /* p += NSLOT * D_MODEL * 4; */                     // 32 MB

    moe_zero_counts<<<1, 32, 0, stream>>>(counts);
    moe_router<<<NTOK / 256, 256, 0, stream>>>(x, gate_w, counts, top_w, etok);
    moe_ffn1<<<NE * 32 * 32, 256, 0, stream>>>(x, w1, b1, counts, etok, Hbuf);
    moe_ffn2<<<NE * 32 * 8, 256, 0, stream>>>(Hbuf, w2, b2, counts, etok, yslot);
    moe_combine<<<NTOK, 256, 0, stream>>>(yslot, top_w, out);
}
